// Model2_6305011990975
// MI455X (gfx1250) — matrix operand order repaired and checked by reading, not compiled, not run on hardware
//
#include <hip/hip_runtime.h>
#include <hip/hip_bf16.h>
#include <cstdint>
#include <cstddef>

// ---------------------------------------------------------------------------
// GRU-D persistent kernel for MI455X (gfx1250, wave32, WMMA bf16)
// B=256 T=128 V=128 H=512 C=10.  16 workgroups x 512 threads (16 waves).
// Each workgroup owns a 16-row batch tile; hidden state lives in LDS; weights
// converted once to bf16 (L2-resident); all GEMMs via v_wmma_f32_16x16x32_bf16.
// ---------------------------------------------------------------------------

typedef __attribute__((ext_vector_type(16))) __bf16 bfx16;
typedef __attribute__((ext_vector_type(8)))  float  fx8;
typedef __attribute__((ext_vector_type(4)))  unsigned int v4u;
typedef __attribute__((ext_vector_type(8)))  int v8i;
typedef __attribute__((ext_vector_type(4)))  int v4i;

#if defined(__AMDGCN__) && __has_builtin(__builtin_amdgcn_tensor_load_to_lds)
#define HAS_TDM 1
#else
#define HAS_TDM 0
#endif

constexpr int B_ = 256, T_ = 128, V_ = 128, H_ = 512, C_ = 10;
constexpr int BT = 16; // batch rows per workgroup

// ---- dynamic LDS layout (bytes) -------------------------------------------
constexpr unsigned OFF_H1   = 0;                        // 16x512 f32
constexpr unsigned OFF_H2   = OFF_H1  + BT*H_*4;
constexpr unsigned OFF_H3   = OFF_H2  + BT*H_*4;
constexpr unsigned OFF_HGF  = OFF_H3  + BT*H_*4;        // decayed h, f32
constexpr unsigned OFF_XH   = OFF_HGF + BT*H_*4;        // x-hat feedback f32
constexpr unsigned OFF_XSTG = OFF_XH  + BT*V_*4;        // TDM x staging f32
constexpr unsigned OFF_YST  = OFF_XSTG+ BT*V_*4;        // logits 16x16 f32
constexpr unsigned OFF_DSM  = OFF_YST + BT*16*4;        // d_t 16 f32
constexpr unsigned OFF_HGB  = OFF_DSM + 64;             // decayed h, bf16
constexpr unsigned OFF_INB  = OFF_HGB + BT*H_*2;        // layer-0 input bf16
constexpr unsigned OFF_BUFA = OFF_INB + BT*V_*2;        // act ping bf16
constexpr unsigned OFF_BUF2 = OFF_BUFA+ BT*H_*2;        // act pong bf16
constexpr unsigned SMEM_BYTES = OFF_BUF2 + BT*H_*2;     // ~197 KB < 320 KB

// ---- bf16 weight workspace layout (elements) ------------------------------
constexpr size_t OWI1 = 0;
constexpr size_t OWH1 = OWI1 + (size_t)3*H_*V_;
constexpr size_t OWI2 = OWH1 + (size_t)3*H_*H_;
constexpr size_t OWH2 = OWI2 + (size_t)3*H_*H_;
constexpr size_t OWI3 = OWH2 + (size_t)3*H_*H_;
constexpr size_t OWH3 = OWI3 + (size_t)3*H_*H_;
constexpr size_t OWX  = OWH3 + (size_t)3*H_*H_;
constexpr size_t OWM  = OWX  + (size_t)V_*H_;
constexpr size_t OWO  = OWM  + (size_t)V_*H_;           // padded 16x512

struct Params {
  const float *x, *d, *m1, *h01, *h02, *h03;
  const float *bi1, *bh1, *bi2, *bh2, *bi3, *bh3;
  const float *bo, *bx, *bm, *Wd, *bd;
  const __bf16 *wi1, *wh1, *wi2, *wh2, *wi3, *wh3, *wx, *wm, *wo;
  float *out;
};

__device__ __forceinline__ float sigmf(float v) { return 1.f / (1.f + __expf(-v)); }

// A fragment (16x32 bf16) from LDS, row-major stride ks.
// Lane layout (ISA 7.12.2): lanes 0-15 row m=lane, K {0-7,16-23}; lanes 16-31 K {8-15,24-31}.
__device__ __forceinline__ bfx16 load_a_lds(const __bf16* s, int ks, int kt, int lane) {
  int m = lane & 15, hf = lane >> 4;
  const __bf16* p = s + m * ks + kt * 32 + hf * 8;
  union { bfx16 v; uint4 q[2]; } u;
  u.q[0] = *(const uint4*)p;
  u.q[1] = *(const uint4*)(p + 16);
  return u.v;
}

// B fragment (32x16 bf16) from global row-major W[N][K] (W row n = output col n).
// Lane n=lane&15 holds column n; lanes<16 K {0-7,16-23}, lanes>=16 K {8-15,24-31}.
__device__ __forceinline__ bfx16 load_b_gl(const __bf16* w, int K, int nt, int kt, int lane) {
  int n = nt * 16 + (lane & 15), hf = lane >> 4;
  const __bf16* p = w + (size_t)n * K + kt * 32 + hf * 8;
  union { bfx16 v; uint4 q[2]; } u;
  u.q[0] = *(const uint4*)p;
  u.q[1] = *(const uint4*)(p + 16);
  return u.v;
}

__device__ __forceinline__ fx8 wmma_bf16(bfx16 a, bfx16 b, fx8 c) {
  return __builtin_amdgcn_wmma_f32_16x16x32_bf16(false, a, false, b, (short)0, c, false, false);
}

// TDM: DMA a [16 x 128] fp32 tile (row stride T_*V_ elems) into LDS.
__device__ __forceinline__ void tdm_load_x_tile(const float* gsrc, unsigned lds_off) {
#if HAS_TDM
  unsigned long long ga = (unsigned long long)(uintptr_t)gsrc;
  v4u g0 = { 1u,                                   // count=1 valid descriptor
             lds_off,                              // lds_addr
             (unsigned)(ga & 0xFFFFFFFFu),
             (unsigned)(((ga >> 32) & 0x01FFFFFFu) | 0x80000000u) }; // type=2
  v8i g1 = { (int)(2u << 16),        // data_size=4B
             (int)(128u << 16),      // tensor_dim0 = 128  (bits 63:48)
             (int)(16u << 16),       // tensor_dim1 = 16   (bits 95:80)
             (int)(128u << 16),      // tile_dim0   = 128  (bits 127:112)
             16,                     // tile_dim1   = 16   (bits 143:128)
             (int)(T_ * V_),         // tensor_dim0_stride (bits 191:160)
             0, 0 };
  v4i z4 = { 0, 0, 0, 0 };
#if __clang_major__ >= 23
  v8i z8 = { 0, 0, 0, 0, 0, 0, 0, 0 };
  __builtin_amdgcn_tensor_load_to_lds(g0, g1, z4, z4, z8, 0);
#else
  __builtin_amdgcn_tensor_load_to_lds(g0, g1, z4, z4, 0);
#endif
#else
  (void)gsrc; (void)lds_off;
#endif
}

__global__ __launch_bounds__(512, 1)
void grud_kernel(Params p) {
  extern __shared__ char smem[];
  float* Hs0  = (float*)(smem + OFF_H1);
  float* Hs1  = (float*)(smem + OFF_H2);
  float* Hs2  = (float*)(smem + OFF_H3);
  float* HGF  = (float*)(smem + OFF_HGF);
  float* XH   = (float*)(smem + OFF_XH);
  float* XSTG = (float*)(smem + OFF_XSTG);
  float* YST  = (float*)(smem + OFF_YST);
  float* DSM  = (float*)(smem + OFF_DSM);
  __bf16* HGB  = (__bf16*)(smem + OFF_HGB);
  __bf16* INB  = (__bf16*)(smem + OFF_INB);
  __bf16* BUFA = (__bf16*)(smem + OFF_BUFA);
  __bf16* BUF2 = (__bf16*)(smem + OFF_BUF2);

  const int tid = threadIdx.x, lane = tid & 31, wave = tid >> 5;
  const int b0 = blockIdx.x * BT;
  float* outY = p.out;
  float* outX = p.out + (size_t)B_ * T_ * C_;
  float* outM = outX + (size_t)B_ * (T_ - 1) * V_;

  // init hidden states from h0_{1,2,3}
  for (int i = tid; i < BT * H_; i += (int)blockDim.x) {
    int r = i >> 9, c = i & (H_ - 1);
    size_t g = (size_t)(b0 + r) * H_ + c;
    Hs0[i] = p.h01[g]; Hs1[i] = p.h02[g]; Hs2[i] = p.h03[g];
  }
  __syncthreads();

  for (int t = 0; t < T_; ++t) {
    // ---- input build: imputation x1 = xh*(1-m) + m*x (raw x at t=0) ----
    for (int i = tid; i < BT * V_; i += (int)blockDim.x) {
      int r = i >> 7, v = i & (V_ - 1);
      float xv = (t == 0 || !HAS_TDM)
                     ? p.x[((size_t)(b0 + r) * T_ + t) * V_ + v]
                     : XSTG[i];
      float x1;
      if (t == 0) {
        x1 = xv;
      } else {
        float mv = p.m1[((size_t)(b0 + r) * T_ + t) * V_ + v];
        x1 = XH[i] * (1.f - mv) + mv * xv;
      }
      INB[i] = (__bf16)x1;
      if (t + 1 < T_)  // warm next step's mask rows (global_prefetch_b8)
        __builtin_prefetch(&p.m1[((size_t)(b0 + r) * T_ + (t + 1)) * V_ + v], 0, 1);
    }
    if (t > 0 && tid < BT) DSM[tid] = p.d[(size_t)(b0 + tid) * T_ + t];
    __syncthreads();

    // async prefetch of next timestep's x tile via Tensor Data Mover
    if (HAS_TDM && wave == 0 && t + 1 < T_)
      tdm_load_x_tile(p.x + ((size_t)b0 * T_ + (t + 1)) * V_, OFF_XSTG);

    // ---- three stacked GRU layers ----
    for (int l = 0; l < 3; ++l) {
      float* Hl = (l == 0) ? Hs0 : (l == 1) ? Hs1 : Hs2;

      // decay prep: hg = h * exp(-relu(Wd*d + bd))  (gamma=1 at t=0)
      for (int i = tid; i < BT * H_; i += (int)blockDim.x) {
        int r = i >> 9, c = i & (H_ - 1);
        float g = 1.f;
        if (t > 0) {
          float a = p.Wd[c] * DSM[r] + p.bd[c];
          g = __expf(-fmaxf(a, 0.f));
        }
        float hg = Hl[i] * g;
        HGF[i] = hg;
        HGB[i] = (__bf16)hg;
      }
      __syncthreads();

      const __bf16* Wi = (l == 0) ? p.wi1 : (l == 1) ? p.wi2 : p.wi3;
      const __bf16* Wh = (l == 0) ? p.wh1 : (l == 1) ? p.wh2 : p.wh3;
      const float*  bi = (l == 0) ? p.bi1 : (l == 1) ? p.bi2 : p.bi3;
      const float*  bh = (l == 0) ? p.bh1 : (l == 1) ? p.bh2 : p.bh3;
      const __bf16* Ain  = (l == 0) ? INB : (l == 1) ? BUFA : BUF2;
      __bf16*       Aout = (l == 1) ? BUF2 : BUFA;
      const int Kin = (l == 0) ? V_ : H_;

      // 32 base N-tiles (r-gate cols), 2 per wave; z at +32 tiles, n at +64.
      for (int jj = 0; jj < 2; ++jj) {
        const int j = wave * 2 + jj;
        fx8 air = {0,0,0,0,0,0,0,0}, aiz = {0,0,0,0,0,0,0,0}, ain_ = {0,0,0,0,0,0,0,0};
        fx8 ahr = {0,0,0,0,0,0,0,0}, ahz = {0,0,0,0,0,0,0,0}, ahn = {0,0,0,0,0,0,0,0};
        for (int kt = 0; kt < Kin / 32; ++kt) {
          bfx16 a = load_a_lds(Ain, Kin, kt, lane);
          air  = wmma_bf16(a, load_b_gl(Wi, Kin, j,      kt, lane), air);
          aiz  = wmma_bf16(a, load_b_gl(Wi, Kin, j + 32, kt, lane), aiz);
          ain_ = wmma_bf16(a, load_b_gl(Wi, Kin, j + 64, kt, lane), ain_);
        }
        for (int kt = 0; kt < H_ / 32; ++kt) {
          bfx16 a = load_a_lds(HGB, H_, kt, lane);
          ahr = wmma_bf16(a, load_b_gl(Wh, H_, j,      kt, lane), ahr);
          ahz = wmma_bf16(a, load_b_gl(Wh, H_, j + 32, kt, lane), ahz);
          ahn = wmma_bf16(a, load_b_gl(Wh, H_, j + 64, kt, lane), ahn);
        }
        // gate math on the 16x16 D tiles (M = r + 8*(lane>=16), N = lane&15)
        const int nc = j * 16 + (lane & 15);
        const float bir = bi[nc], biz = bi[nc + H_], bin = bi[nc + 2 * H_];
        const float bhr = bh[nc], bhz = bh[nc + H_], bhn = bh[nc + 2 * H_];
        #pragma unroll
        for (int rr = 0; rr < 8; ++rr) {
          int mr = (lane >> 4) * 8 + rr;
          float rg = sigmf(air[rr] + bir + ahr[rr] + bhr);
          float zg = sigmf(aiz[rr] + biz + ahz[rr] + bhz);
          float ng = tanhf(ain_[rr] + bin + rg * (ahn[rr] + bhn));
          float hd = HGF[mr * H_ + nc];
          float hv = (1.f - zg) * ng + zg * hd;
          Hl[mr * H_ + nc]   = hv;
          Aout[mr * H_ + nc] = (__bf16)hv;
        }
      }
      __syncthreads();
    }

    // ---- heads: c bf16 in BUFA. waves 0-7 -> Wx, 8-15 -> Wm, wave 0 also Wo.
    {
      const bool isX = (wave < 8);
      const int  nt  = isX ? wave : wave - 8;
      const __bf16* W = isX ? p.wx : p.wm;
      fx8 acc = {0,0,0,0,0,0,0,0};
      for (int kt = 0; kt < H_ / 32; ++kt) {
        bfx16 a = load_a_lds(BUFA, H_, kt, lane);
        acc = wmma_bf16(a, load_b_gl(W, H_, nt, kt, lane), acc);
      }
      const int col = nt * 16 + (lane & 15);
      const float bb = isX ? p.bx[col] : p.bm[col];
      #pragma unroll
      for (int rr = 0; rr < 8; ++rr) {
        int mr = (lane >> 4) * 8 + rr;
        float v = acc[rr] + bb;
        if (isX) {
          XH[mr * V_ + col] = v;  // feedback for next step's imputation
          if (t > 0) outX[((size_t)(b0 + mr) * (T_ - 1) + (t - 1)) * V_ + col] = v;
        } else if (t > 0) {
          outM[((size_t)(b0 + mr) * (T_ - 1) + (t - 1)) * V_ + col] = sigmf(v);
        }
      }
      if (wave == 0) {  // class logits (Wo padded to 16 rows)
        fx8 ao = {0,0,0,0,0,0,0,0};
        for (int kt = 0; kt < H_ / 32; ++kt) {
          bfx16 a = load_a_lds(BUFA, H_, kt, lane);
          ao = wmma_bf16(a, load_b_gl(p.wo, H_, 0, kt, lane), ao);
        }
        const int c = lane & 15;
        #pragma unroll
        for (int rr = 0; rr < 8; ++rr) {
          int mr = (lane >> 4) * 8 + rr;
          YST[mr * 16 + c] = ao[rr] + (c < C_ ? p.bo[c] : 0.f);
        }
      }
    }
#if HAS_TDM && __has_builtin(__builtin_amdgcn_s_wait_tensorcnt)
    if (wave == 0 && t + 1 < T_) __builtin_amdgcn_s_wait_tensorcnt(0);
#endif
    __syncthreads();

    // softmax over C=10, one row per thread
    if (tid < BT) {
      float mx = -1e30f;
      for (int c = 0; c < C_; ++c) mx = fmaxf(mx, YST[tid * 16 + c]);
      float e[C_], s = 0.f;
      for (int c = 0; c < C_; ++c) { e[c] = __expf(YST[tid * 16 + c] - mx); s += e[c]; }
      float inv = 1.f / s;
      for (int c = 0; c < C_; ++c)
        outY[((size_t)(b0 + tid) * T_ + t) * C_ + c] = e[c] * inv;
    }
    __syncthreads();
  }
}

// ---------------------------------------------------------------------------
// weight prep: fp32 -> bf16 (row-major kept; B-fragment loads need no repack)
// ---------------------------------------------------------------------------
__global__ void cvt_weights(const float* __restrict__ s, __bf16* __restrict__ d, int n) {
  for (int i = blockIdx.x * blockDim.x + threadIdx.x; i < n; i += gridDim.x * blockDim.x)
    d[i] = (__bf16)s[i];
}
__global__ void cvt_wo_pad(const float* __restrict__ wo, __bf16* __restrict__ d) {
  int i = blockIdx.x * blockDim.x + threadIdx.x;
  if (i < 16 * H_) {
    int r = i >> 9;
    d[i] = (r < C_) ? (__bf16)wo[i] : (__bf16)0.f;
  }
}

extern "C" void kernel_launch(void* const* d_in, const int* in_sizes, int n_in,
                              void* d_out, int out_size, void* d_ws, size_t ws_size,
                              hipStream_t stream) {
  (void)in_sizes; (void)n_in; (void)out_size; (void)ws_size;
  const float* x    = (const float*)d_in[0];
  const float* d    = (const float*)d_in[1];
  const float* m1   = (const float*)d_in[2];
  const float* h01  = (const float*)d_in[3];
  const float* h02  = (const float*)d_in[4];
  const float* h03  = (const float*)d_in[5];
  const float* Wi1  = (const float*)d_in[6];
  const float* Wh1  = (const float*)d_in[7];
  const float* bi1  = (const float*)d_in[8];
  const float* bh1  = (const float*)d_in[9];
  const float* Wi2  = (const float*)d_in[10];
  const float* Wh2  = (const float*)d_in[11];
  const float* bi2  = (const float*)d_in[12];
  const float* bh2  = (const float*)d_in[13];
  const float* Wi3  = (const float*)d_in[14];
  const float* Wh3  = (const float*)d_in[15];
  const float* bi3  = (const float*)d_in[16];
  const float* bh3  = (const float*)d_in[17];
  const float* Wo   = (const float*)d_in[18];
  const float* bo   = (const float*)d_in[19];
  const float* Wx   = (const float*)d_in[20];
  const float* bx   = (const float*)d_in[21];
  const float* Wm   = (const float*)d_in[22];
  const float* bm   = (const float*)d_in[23];
  const float* Wd   = (const float*)d_in[24];
  const float* bd   = (const float*)d_in[25];

  __bf16* ws = (__bf16*)d_ws;
  auto cv = [&](const float* s, size_t off, int n) {
    cvt_weights<<<(n + 255) / 256, 256, 0, stream>>>(s, ws + off, n);
  };
  cv(Wi1, OWI1, 3 * H_ * V_);
  cv(Wh1, OWH1, 3 * H_ * H_);
  cv(Wi2, OWI2, 3 * H_ * H_);
  cv(Wh2, OWH2, 3 * H_ * H_);
  cv(Wi3, OWI3, 3 * H_ * H_);
  cv(Wh3, OWH3, 3 * H_ * H_);
  cv(Wx,  OWX,  V_ * H_);
  cv(Wm,  OWM,  V_ * H_);
  cvt_wo_pad<<<(16 * H_ + 255) / 256, 256, 0, stream>>>(Wo, ws + OWO);

  Params p;
  p.x = x; p.d = d; p.m1 = m1; p.h01 = h01; p.h02 = h02; p.h03 = h03;
  p.bi1 = bi1; p.bh1 = bh1; p.bi2 = bi2; p.bh2 = bh2; p.bi3 = bi3; p.bh3 = bh3;
  p.bo = bo; p.bx = bx; p.bm = bm; p.Wd = Wd; p.bd = bd;
  p.wi1 = ws + OWI1; p.wh1 = ws + OWH1; p.wi2 = ws + OWI2; p.wh2 = ws + OWH2;
  p.wi3 = ws + OWI3; p.wh3 = ws + OWH3; p.wx = ws + OWX; p.wm = ws + OWM; p.wo = ws + OWO;
  p.out = (float*)d_out;

  (void)hipFuncSetAttribute((const void*)grud_kernel,
                            hipFuncAttributeMaxDynamicSharedMemorySize,
                            (int)SMEM_BYTES);
  grud_kernel<<<B_ / BT, 512, SMEM_BYTES, stream>>>(p);
}
